// MCARResnet_34462817583149
// MI455X (gfx1250) — matrix operand order repaired and checked by reading, not compiled, not run on hardware
//
#include <hip/hip_runtime.h>
#include <math.h>

typedef __attribute__((ext_vector_type(16))) __bf16 bfvec16;
typedef __attribute__((ext_vector_type(8)))  float  fvec8;

#define TOPN 4
#define NC   20
#define BB   8

// ---------------------------------------------------------------------------
// Implicit-GEMM convolution on the WMMA path.
// Block: 128 threads = 4 waves. Each wave owns 16 output channels (64/block),
// block shares a 32-pixel tile -> each wave does TWO v_wmma_f32_16x16x32_bf16
// per K-chunk with one shared A fragment. Staging is branchless (clamped
// addresses + cndmask zeroing), loads batched into registers before LDS store.
// KS is a template parameter so kk/(KS*KS) is division-by-constant; all
// spatial sizes are powers of two and passed as log2 shifts.
// ---------------------------------------------------------------------------
template<int KS>
__global__ __launch_bounds__(128) void conv_bf16_wmma(
    const float* __restrict__ in, const float* __restrict__ wgt,
    float* __restrict__ out,
    int Cin, int Cout, int wsh, int owsh,   // input/out width = 1<<wsh / 1<<owsh (square)
    int stride, int pad,
    const float* __restrict__ bn_g, const float* __restrict__ bn_b,
    const float* __restrict__ bn_m, const float* __restrict__ bn_v,
    int relu)
{
    constexpr int KK2 = KS * KS;
    __shared__ __align__(32) __bf16 ldsA[4 * 512];   // per-wave 16x32 weight tile
    __shared__ __align__(32) __bf16 ldsB[2 * 512];   // two shared 32x16 im2col tiles

    const int tid  = threadIdx.x;
    const int lane = tid & 31;
    const int wave = tid >> 5;
    const int img  = blockIdx.z;
    const int W_   = 1 << wsh;
    const int OW_  = 1 << owsh;
    const int coutBase = blockIdx.y * 64 + wave * 16;
    const int pixBase  = blockIdx.x * 32;
    const int K = Cin * KK2;

    const float* inImg  = in  + (((size_t)img * Cin)  << (2 * wsh));
    float*       outImg = out + (((size_t)img * Cout) << (2 * owsh));

    // ---- loop-invariant B-staging coords: this thread always feeds pixel p
    const int p_n   = tid & 31;                 // pixel within tile
    const int p_pix = pixBase + p_n;
    const int p_oy  = p_pix >> owsh;
    const int p_ox  = p_pix & (OW_ - 1);
    const int p_iy0 = p_oy * stride - pad;
    const int p_ix0 = p_ox * stride - pad;
    const int b_tile = p_n >> 4;
    const int b_nn   = p_n & 15;
    const int b_kb   = tid >> 5;                // k residue (mod 4)

    // ---- loop-invariant A-staging coords: this lane always feeds cout row m
    const int a_m  = lane & 15;
    const int a_kb = lane >> 4;                 // k parity
    const float* wRowM = wgt + (size_t)(coutBase + a_m) * K;

    fvec8 acc0 = {};
    fvec8 acc1 = {};

    for (int k0 = 0; k0 < K; k0 += 32) {
        // ---- A: load 16 weights unconditionally (clamped), zero via select
        float av[16];
        #pragma unroll
        for (int t = 0; t < 16; ++t) {
            int kk = k0 + a_kb + 2 * t;
            int kc = kk < K ? kk : K - 1;
            float v = wRowM[kc];
            av[t] = kk < K ? v : 0.f;
        }
        #pragma unroll
        for (int t = 0; t < 16; ++t) {
            int k  = a_kb + 2 * t;
            int fl = a_m + 16 * ((k >> 3) & 1);
            int fi = (k & 7) + 8 * ((k >> 4) & 1);
            ldsA[wave * 512 + fl * 16 + fi] = (__bf16)av[t];
        }
        // ---- B: branchless im2col gather, loads batched before LDS stores
        float bv[8];
        #pragma unroll
        for (int t = 0; t < 8; ++t) {
            int kk = k0 + b_kb + 4 * t;
            int kc = kk < K ? kk : K - 1;
            int cin = kc / KK2;                 // division by constant
            int rem = kc - cin * KK2;
            int dy  = rem / KS;
            int dx  = rem - dy * KS;
            int iy  = p_iy0 + dy;
            int ix  = p_ix0 + dx;
            bool ok = (kk < K) && (iy >= 0) && (iy < W_) && (ix >= 0) && (ix < W_);
            int iyc = min(max(iy, 0), W_ - 1);
            int ixc = min(max(ix, 0), W_ - 1);
            float v = inImg[(((size_t)cin) << (2 * wsh)) + ((size_t)iyc << wsh) + ixc];
            bv[t] = ok ? v : 0.f;
        }
        #pragma unroll
        for (int t = 0; t < 8; ++t) {
            int k  = b_kb + 4 * t;
            int fl = b_nn + 16 * ((k >> 3) & 1);
            int fi = (k & 7) + 8 * ((k >> 4) & 1);
            ldsB[b_tile * 512 + fl * 16 + fi] = (__bf16)bv[t];
        }
        // pull next weight chunk toward cache while the matrix unit works
        if (k0 + 32 < K)
            __builtin_prefetch(wRowM + k0 + 32, 0, 1);
        __syncthreads();

        bfvec16 a  = *(const bfvec16*)(&ldsA[wave * 512 + lane * 16]);
        bfvec16 b0 = *(const bfvec16*)(&ldsB[lane * 16]);
        bfvec16 b1 = *(const bfvec16*)(&ldsB[512 + lane * 16]);
        acc0 = __builtin_amdgcn_wmma_f32_16x16x32_bf16(
                   false, a, false, b0, (short)0, acc0, false, false);
        acc1 = __builtin_amdgcn_wmma_f32_16x16x32_bf16(
                   false, a, false, b1, (short)0, acc1, false, false);
        __syncthreads();
    }

    // D fragment: VGPR r -> M=r (lanes 0-15) / M=r+8 (lanes 16-31), N=lane&15
    const int n    = lane & 15;
    const int mOff = (lane >> 4) * 8;
    #pragma unroll
    for (int r = 0; r < 8; ++r) {
        int c = coutBase + mOff + r;
        float v0 = acc0[r], v1 = acc1[r];
        if (bn_g) {
            float inv = bn_g[c] * rsqrtf(bn_v[c] + 1e-5f);
            float sh  = bn_b[c] - bn_m[c] * inv;
            v0 = v0 * inv + sh;
            v1 = v1 * inv + sh;
        }
        if (relu) { v0 = fmaxf(v0, 0.f); v1 = fmaxf(v1, 0.f); }
        size_t rowOff = ((size_t)c) << (2 * owsh);
        outImg[rowOff + pixBase + n]      = v0;
        outImg[rowOff + pixBase + 16 + n] = v1;
    }
}

// ---------------------------------------------------------------------------
// Small glue kernels (latency-trivial)
// ---------------------------------------------------------------------------
__global__ void pool_mean64(const float* __restrict__ act, float* __restrict__ pooled,
                            int total)
{
    int i = blockIdx.x * blockDim.x + threadIdx.x;
    if (i >= total) return;
    const float* a = act + (size_t)i * 64;
    float s = 0.f;
    #pragma unroll
    for (int j = 0; j < 64; ++j) s += a[j];
    pooled[i] = s * (1.f / 64.f);
}

__global__ void classify_sigmoid(const float* __restrict__ pooled,
                                 const float* __restrict__ cw,
                                 const float* __restrict__ cb,
                                 float* __restrict__ outv, int total)
{
    int i = blockIdx.x * blockDim.x + threadIdx.x;
    if (i >= total) return;
    int b = i / NC, o = i - b * NC;
    const float* pv = pooled + (size_t)b * 1024;
    const float* wv = cw + (size_t)o * 1024;
    float s = cb[o];
    for (int c = 0; c < 1024; ++c) s += pv[c] * wv[c];
    outv[i] = 1.f / (1.f + expf(-s));
}

__global__ void cam_sigmoid(const float* __restrict__ act,
                            const float* __restrict__ cw,
                            const float* __restrict__ cb,
                            float* __restrict__ cam, int total)
{
    int i = blockIdx.x * blockDim.x + threadIdx.x;
    if (i >= total) return;
    int p  = i & 63;
    int bo = i >> 6;
    int o  = bo % NC;
    int b  = bo / NC;
    const float* a  = act + (size_t)b * 1024 * 64 + p;
    const float* wv = cw + (size_t)o * 1024;
    float s = cb[o];
    for (int c = 0; c < 1024; ++c) s += a[(size_t)c * 64] * wv[c];
    cam[i] = 1.f / (1.f + expf(-s));
}

__global__ void resize_8to128(const float* __restrict__ cam, float* __restrict__ cs,
                              int total)
{
    int i = blockIdx.x * blockDim.x + threadIdx.x;
    if (i >= total) return;
    int ox = i & 127, oy = (i >> 7) & 127, bo = i >> 14;
    float fy = oy * (7.f / 127.f);
    float fx = ox * (7.f / 127.f);
    int y0 = (int)floorf(fy), x0 = (int)floorf(fx);
    float wy = fy - (float)y0, wx = fx - (float)x0;
    y0 = min(max(y0, 0), 7); x0 = min(max(x0, 0), 7);
    int y1 = min(y0 + 1, 7), x1 = min(x0 + 1, 7);
    const float* c = cam + (size_t)bo * 64;
    float v = (1.f - wy) * ((1.f - wx) * c[y0 * 8 + x0] + wx * c[y0 * 8 + x1])
            +        wy  * ((1.f - wx) * c[y1 * 8 + x0] + wx * c[y1 * 8 + x1]);
    cs[i] = v;
}

__global__ void row_col_max(const float* __restrict__ cs,
                            float* __restrict__ wsc, float* __restrict__ hsc, int total)
{
    int i = blockIdx.x * blockDim.x + threadIdx.x;
    if (i >= total) return;
    int q  = i & 127;
    int bo = i >> 7;
    const float* base = cs + (size_t)bo * 16384;
    float mw = base[q], mh = base[q * 128];
    for (int j = 1; j < 128; ++j) {
        mw = fmaxf(mw, base[j * 128 + q]);   // max over rows (axis 2) -> wscore
        mh = fmaxf(mh, base[q * 128 + j]);   // max over cols (axis 3) -> hscore
    }
    wsc[i] = mw;
    hsc[i] = mh;
}

__global__ void topk4(const float* __restrict__ gs, int* __restrict__ top)
{
    int b = blockIdx.x * blockDim.x + threadIdx.x;
    if (b >= BB) return;
    unsigned used = 0;
    for (int t = 0; t < TOPN; ++t) {
        int bi = 0; float bv = -3.0e38f;
        for (int o = 0; o < NC; ++o)
            if (!((used >> o) & 1u) && gs[b * NC + o] > bv) { bv = gs[b * NC + o]; bi = o; }
        used |= 1u << bi;
        top[b * TOPN + t] = bi;
    }
}

__global__ void gather_norm(const float* __restrict__ wsc, const float* __restrict__ hsc,
                            const int* __restrict__ top,
                            float* __restrict__ xs, float* __restrict__ ys)
{
    int j = blockIdx.x * blockDim.x + threadIdx.x;
    if (j >= BB * TOPN) return;
    int b = j >> 2;
    int o = top[j];
    for (int which = 0; which < 2; ++which) {
        const float* src = (which ? hsc : wsc) + ((size_t)b * NC + o) * 128;
        float* dst = (which ? ys : xs) + (size_t)j * 128;
        float mn = src[0], mx = src[0];
        for (int i = 1; i < 128; ++i) { mn = fminf(mn, src[i]); mx = fmaxf(mx, src[i]); }
        if (mx == mn) {
            float d = (mx == 0.f) ? 1.f : mx;
            for (int i = 0; i < 128; ++i) dst[i] = src[i] / d;
        } else {
            float inv = 1.f / (mx - mn);
            for (int i = 0; i < 128; ++i) dst[i] = (src[i] - mn) * inv;
        }
    }
}

__device__ void obj_loc_dev(const float* s, int* pz1, int* pz2)
{
    const int S = 128, minsize = 16;   // ceil(128*0.125)
    int pos[128];
    int ncross = 0;
    int prev = (s[0] > 0.5f) - (s[0] < 0.5f);
    for (int i = 0; i < S - 1; ++i) {
        int cur = (s[i + 1] > 0.5f) - (s[i + 1] < 0.5f);
        int d = cur - prev;
        if (d == 2 || d == -2) pos[ncross++] = i;
        prev = cur;
    }
    int sa[129], sb[129];
    float smax[129];
    int nseg = ncross + 1;
    sa[0] = 0;
    for (int j = 0; j < ncross; ++j) { sb[j] = pos[j]; sa[j + 1] = pos[j]; }
    sb[ncross] = S;
    float m = -3.0e38f;
    for (int j = 0; j < nseg; ++j) {
        float mx = -3.0e38f;
        for (int i = sa[j]; i < sb[j]; ++i) mx = fmaxf(mx, s[i]);
        smax[j] = mx;
        m = fmaxf(m, mx);
    }
    int best = 0, bestval = -2;
    for (int j = 0; j < nseg; ++j) {
        int val = (smax[j] == m) ? (sb[j] - sa[j]) : -1;
        if (val > bestval) { bestval = val; best = j; }
    }
    int zmin = sa[best], zmax = sb[best];
    int need = (zmax - zmin) <= minsize;
    int padv = minsize - (zmax - zmin);
    int cp = (padv + 1) >> 1;
    if (need && zmin > cp && (S - zmax) > padv) { zmin = zmin - cp + 1; zmax = zmax + cp; }
    if (need && zmin < cp)                      { zmin = 0; zmax = minsize; }
    if (need && (S - zmax) < cp)                { zmin = S - minsize + 1; zmax = S; }
    if (ncross == 0)                            { zmin = minsize; zmax = 112; } // ceil(128*.875)
    *pz1 = zmin; *pz2 = zmax;
}

__global__ void objloc_kernel(const float* __restrict__ xs, const float* __restrict__ ys,
                              int* bx1, int* bx2, int* by1, int* by2)
{
    int r = blockIdx.x * blockDim.x + threadIdx.x;
    if (r < 32) {
        obj_loc_dev(xs + (size_t)r * 128, &bx1[r], &bx2[r]);
    } else if (r < 64) {
        int j = r - 32;
        obj_loc_dev(ys + (size_t)j * 128, &by1[j], &by2[j]);
    }
}

__global__ void crop_resize(const float* __restrict__ x,
                            const int* __restrict__ bx1, const int* __restrict__ bx2,
                            const int* __restrict__ by1, const int* __restrict__ by2,
                            float* __restrict__ lin, int total)
{
    int i = blockIdx.x * blockDim.x + threadIdx.x;
    if (i >= total) return;
    int ox = i & 127;
    int oy = (i >> 7) & 127;
    int c  = (i >> 14) % 3;
    int j  = i / 49152;          // crop id (0..31)
    int b  = j >> 2;             // source image
    float fy = (float)by1[j] + oy * (float)(by2[j] - 1 - by1[j]) * (1.f / 127.f);
    float fx = (float)bx1[j] + ox * (float)(bx2[j] - 1 - bx1[j]) * (1.f / 127.f);
    float y0f = floorf(fy), x0f = floorf(fx);
    float wy = fy - y0f, wx = fx - x0f;
    int y0 = min(max((int)y0f, 0), 127);
    int x0 = min(max((int)x0f, 0), 127);
    int y1 = min(y0 + 1, 127);
    int x1 = min(x0 + 1, 127);
    const float* img = x + ((size_t)b * 3 + c) * 16384;
    float v = (1.f - wy) * ((1.f - wx) * img[y0 * 128 + x0] + wx * img[y0 * 128 + x1])
            +        wy  * ((1.f - wx) * img[y1 * 128 + x0] + wx * img[y1 * 128 + x1]);
    lin[i] = v;
}

__global__ void local_max(const float* __restrict__ lg, float* __restrict__ ls)
{
    int i = blockIdx.x * blockDim.x + threadIdx.x;
    if (i >= BB * NC) return;
    int b = i / NC, o = i - b * NC;
    float m = lg[(size_t)(b * TOPN) * NC + o];
    for (int t = 1; t < TOPN; ++t) m = fmaxf(m, lg[(size_t)(b * TOPN + t) * NC + o]);
    ls[i] = m;
}

// ---------------------------------------------------------------------------
// Host side
// ---------------------------------------------------------------------------
static void run_features(const float* img, const float* c1w,
                         const float* g, const float* be, const float* mu, const float* va,
                         const float* w1, const float* w2, const float* w3, const float* w4,
                         float* bufA, float* bufB, hipStream_t s)
{
    dim3 blk(128, 1, 1);
    // conv1 7x7 s2 p3, 3->64, 128->64, fused BN, no relu
    conv_bf16_wmma<7><<<dim3(128, 1, 8), blk, 0, s>>>(img,  c1w, bufA, 3,   64,   7, 6, 2, 3, g, be, mu, va, 0);
    // w1 3x3 s1, 64->128, 64x64
    conv_bf16_wmma<3><<<dim3(128, 2, 8), blk, 0, s>>>(bufA, w1,  bufB, 64,  128,  6, 6, 1, 1, nullptr, nullptr, nullptr, nullptr, 1);
    // w2 3x3 s2, 128->256, ->32x32
    conv_bf16_wmma<3><<<dim3(32,  4, 8), blk, 0, s>>>(bufB, w2,  bufA, 128, 256,  6, 5, 2, 1, nullptr, nullptr, nullptr, nullptr, 1);
    // w3 3x3 s2, 256->512, ->16x16
    conv_bf16_wmma<3><<<dim3(8,   8, 8), blk, 0, s>>>(bufA, w3,  bufB, 256, 512,  5, 4, 2, 1, nullptr, nullptr, nullptr, nullptr, 1);
    // w4 3x3 s2, 512->1024, ->8x8
    conv_bf16_wmma<3><<<dim3(2,  16, 8), blk, 0, s>>>(bufB, w4,  bufA, 512, 1024, 4, 3, 2, 1, nullptr, nullptr, nullptr, nullptr, 1);
    // final activations: bufA = (8, 1024, 8, 8)
}

extern "C" void kernel_launch(void* const* d_in, const int* in_sizes, int n_in,
                              void* d_out, int out_size, void* d_ws, size_t ws_size,
                              hipStream_t stream)
{
    (void)in_sizes; (void)n_in; (void)out_size; (void)ws_size;
    const float* x     = (const float*)d_in[0];
    const float* c1w   = (const float*)d_in[1];
    const float* bn_g  = (const float*)d_in[2];
    const float* bn_b  = (const float*)d_in[3];
    const float* bn_m  = (const float*)d_in[4];
    const float* bn_v  = (const float*)d_in[5];
    const float* w1    = (const float*)d_in[6];
    const float* w2    = (const float*)d_in[7];
    const float* w3    = (const float*)d_in[8];
    const float* w4    = (const float*)d_in[9];
    const float* cls_w = (const float*)d_in[10];
    const float* cls_b = (const float*)d_in[11];

    float* out = (float*)d_out;
    float* o_global = out;                 // (8,20)
    float* o_local  = out + 160;           // (8,20)
    float* o_cam    = out + 320;           // (8,20,128,128)
    float* o_w      = o_cam + 2621440;     // (8,20,128)
    float* o_h      = o_w + 20480;         // (8,20,128)
    float* o_lin    = o_h + 20480;         // (32,3,128,128)

    float* ws      = (float*)d_ws;
    float* bufA    = ws;                    // 2,097,152 floats
    float* bufB    = bufA + 2097152;        // 4,194,304 floats
    float* pooled  = bufB + 4194304;        // 8*1024
    float* pooledL = pooled + 8192;         // 32*1024
    float* cam     = pooledL + 32768;       // 8*20*64
    float* xs      = cam + 10240;           // 32*128
    float* ys      = xs + 4096;             // 32*128
    float* logitsL = ys + 4096;             // 32*20
    int*   top     = (int*)(logitsL + 640); // 32
    int*   bx1     = top + 32;
    int*   bx2     = bx1 + 32;
    int*   by1     = bx2 + 32;
    int*   by2     = by1 + 32;

    // ---------------- global stream ----------------
    run_features(x, c1w, bn_g, bn_b, bn_m, bn_v, w1, w2, w3, w4, bufA, bufB, stream);
    pool_mean64<<<32, 256, 0, stream>>>(bufA, pooled, 8 * 1024);
    classify_sigmoid<<<1, 256, 0, stream>>>(pooled, cls_w, cls_b, o_global, 160);
    cam_sigmoid<<<40, 256, 0, stream>>>(bufA, cls_w, cls_b, cam, 10240);
    resize_8to128<<<10240, 256, 0, stream>>>(cam, o_cam, 2621440);
    row_col_max<<<80, 256, 0, stream>>>(o_cam, o_w, o_h, 20480);
    topk4<<<1, 32, 0, stream>>>(o_global, top);
    gather_norm<<<1, 32, 0, stream>>>(o_w, o_h, top, xs, ys);
    objloc_kernel<<<1, 64, 0, stream>>>(xs, ys, bx1, bx2, by1, by2);
    crop_resize<<<6144, 256, 0, stream>>>(x, bx1, bx2, by1, by2, o_lin, 1572864);

    // ---------------- local stream: 32 crops in 4 chunks of 8 ----------------
    for (int ci = 0; ci < 4; ++ci) {
        run_features(o_lin + (size_t)ci * 8 * 3 * 16384, c1w, bn_g, bn_b, bn_m, bn_v,
                     w1, w2, w3, w4, bufA, bufB, stream);
        pool_mean64<<<32, 256, 0, stream>>>(bufA, pooledL + (size_t)ci * 8 * 1024, 8 * 1024);
    }
    classify_sigmoid<<<3, 256, 0, stream>>>(pooledL, cls_w, cls_b, logitsL, 640);
    local_max<<<1, 160, 0, stream>>>(logitsL, o_local);
}
